// PhysNetPretrain_6863357739298
// MI455X (gfx1250) — compile-verified, matrix operand order checked by reading, not run on hardware
//
#include <hip/hip_runtime.h>
#include <hip/hip_bf16.h>
#include <math.h>

// ---------------------------------------------------------------------------
// PhysNet pretrain forward, fused for gfx1250 (MI455X).
// Per block: node-level di/dj GEMMs (hoisted from edges), then a single fused
// per-edge kernel running RBF gating + 9 chained [32x128]x[128x128] GEMMs on
// v_wmma_f32_16x16x32_f16 with weights resident in VGPR B-fragments, then
// atomic scatter-add.  Gathers for tile t+1 are issued into registers while
// tile t's GEMM chain runs (cross-tile software pipeline), so the workgroup
// never stalls on the random-access gather latency.
// ---------------------------------------------------------------------------

typedef __attribute__((ext_vector_type(16))) _Float16 v16h;
typedef __attribute__((ext_vector_type(8)))  float    v8f;

#define FDIM   128
#define TM     32    // edge rows per tile
#define FPS    132   // f32 LDS row stride (bank rotation)
#define APH    136   // f16 LDS row stride (bank rotation)
#define LOG2F_ 0.69314718056f
#define WIDTH2 6.2505675f          // (0.5 / ((1-e^-10)/5))^2
#define EXPM10 4.5399930e-05f      // e^-10

__device__ __forceinline__ float sspf(float v) {
  // shifted softplus: log(1+e^v) - log(2), overflow-safe
  return fmaxf(v, 0.f) + __logf(1.f + __expf(-fabsf(v))) - LOG2F_;
}

// --- WMMA fragment loaders (gfx1250 16-bit layouts, wave32) ----------------
// A (16x32, MxK): lane l -> row M=l&15; koff = (l>=16)?8:0;
//   a[0..7] = K=kbase+koff+0..7 ; a[8..15] = K=kbase+16+koff+0..7
__device__ __forceinline__ v16h load_afrag_lds(const _Float16* A, int kbase, int lane) {
  const int m    = lane & 15;
  const int koff = (lane >> 4) << 3;
  const _Float16* r0 = A + m * APH + kbase + koff;
  v16h a;
#pragma unroll
  for (int j = 0; j < 8; ++j) a[j] = r0[j];
#pragma unroll
  for (int j = 0; j < 8; ++j) a[8 + j] = r0[16 + j];
  return a;
}

// B (32x16, KxN) from weight W[dout=128][din=128] row-major: B[k][n] = W[n][k]
__device__ __forceinline__ v16h load_bfrag_w(const float* __restrict__ W,
                                             int ncol, int kbase, int lane) {
  const int n    = ncol + (lane & 15);
  const int koff = (lane >> 4) << 3;
  const float* r0 = W + (size_t)n * FDIM + kbase + koff;
  v16h b;
#pragma unroll
  for (int j = 0; j < 8; ++j) b[j] = (_Float16)r0[j];
#pragma unroll
  for (int j = 0; j < 8; ++j) b[8 + j] = (_Float16)r0[16 + j];
  return b;
}

// full K=128 GEMM for one 16x16 output tile: C = A(16x128) x B + binit
__device__ __forceinline__ v8f gemm_tile(const _Float16* A, const v16h* B,
                                         float binit, int lane) {
  v8f c;
#pragma unroll
  for (int r = 0; r < 8; ++r) c[r] = binit;
#pragma unroll
  for (int kc = 0; kc < 4; ++kc) {
    v16h a = load_afrag_lds(A, kc * 32, lane);
    c = __builtin_amdgcn_wmma_f32_16x16x32_f16(false, a, false, B[kc],
                                               (short)0, c, false, false);
  }
  return c;
}

// one residual stage on a 32-row tile:
//   m += W2 @ (W1 @ ssp(m) + b1) + b2   (in-place on mS)
__device__ __forceinline__ void residual_stage32(float* mS, _Float16* aF, _Float16* hF,
                                                 const v16h* B1, const v16h* B2,
                                                 float b1, float b2,
                                                 int t, int lane, int ncol,
                                                 int nl, int mrow0) {
  // stage A = f16(ssp(m)) over all 32 rows
  {
    const int er = t >> 3;
    const int f0 = (t & 7) << 4;
#pragma unroll
    for (int j = 0; j < 16; ++j)
      aF[er * APH + f0 + j] = (_Float16)sspf(mS[er * FPS + f0 + j]);
  }
  __syncthreads();
  // h = ssp(m) @ W1^T + b1  -> hF (f16 intermediate), both subtiles
#pragma unroll
  for (int sub = 0; sub < 2; ++sub) {
    v8f c1 = gemm_tile(aF + sub * 16 * APH, B1, b1, lane);
#pragma unroll
    for (int r = 0; r < 8; ++r)
      hF[(sub * 16 + mrow0 + r) * APH + ncol + nl] = (_Float16)c1[r];
  }
  __syncthreads();
  // m += h @ W2^T + b2, both subtiles
#pragma unroll
  for (int sub = 0; sub < 2; ++sub) {
    v8f c2 = gemm_tile(hF + sub * 16 * APH, B2, b2, lane);
#pragma unroll
    for (int r = 0; r < 8; ++r)
      mS[(sub * 16 + mrow0 + r) * FPS + ncol + nl] += c2[r];
  }
  __syncthreads();
}

// ---------------------------------------------------------------------------
struct EdgeParams {
  const float *k2fW;
  const float *r0d1W, *r0d1b, *r0d2W, *r0d2b;
  const float *r1d1W, *r1d1b, *r1d2W, *r1d2b;
  const float *denseW, *denseb, *u;
};

__global__ __launch_bounds__(256, 1)
void edge_kernel(const float* __restrict__ aiN, const float* __restrict__ ajN,
                 const int* __restrict__ src, const int* __restrict__ dst,
                 const float* __restrict__ R, EdgeParams P,
                 float* __restrict__ accum, int E) {
  __shared__ float    mS [TM * FPS];
  __shared__ float    aiS[TM * FPS];
  __shared__ _Float16 aF [TM * APH];
  __shared__ _Float16 hF [TM * APH];
  __shared__ float    rbfS[2 * TM * 5];   // ping-pong across pipelined tiles
  __shared__ float    k2fS[FDIM * 5];

  const int t     = threadIdx.x;
  const int lane  = t & 31;
  const int wave  = t >> 5;         // 8 waves: wave owns columns [16w,16w+16)
  const int ncol  = wave << 4;
  const int nl    = lane & 15;
  const int mrow0 = (lane >> 4) << 3;
  const int er    = t >> 3;         // this thread's edge row (32 rows, 8 thr/row)
  const int f0    = (t & 7) << 4;   // 16 features per thread

  for (int i = t; i < FDIM * 5; i += 256) k2fS[i] = P.k2fW[i];

  // resident weight fragments: 5 GEMMs x 4 K-chunks, per-wave N-tile
  v16h Bf[5][4];
  {
    const float* Ws[5] = {P.r0d1W, P.r0d2W, P.r1d1W, P.r1d2W, P.denseW};
#pragma unroll
    for (int g = 0; g < 5; ++g)
#pragma unroll
      for (int kc = 0; kc < 4; ++kc)
        Bf[g][kc] = load_bfrag_w(Ws[g], ncol, kc * 32, lane);
  }
  const float bs1[2] = {P.r0d1b[ncol + nl], P.r1d1b[ncol + nl]};
  const float bs2[2] = {P.r0d2b[ncol + nl], P.r1d2b[ncol + nl]};
  const float bden   = P.denseb[ncol + nl];
  const float ul     = P.u[ncol + nl];

  const int ntiles = (E + TM - 1) / TM;

  // ---- pipeline prologue: stage gather + RBF for the first tile ----
  float aiv[16], ajv[16];
  int   nextD;
  {
    const int tile0 = blockIdx.x;
    if (t < TM) {  // RBF for tile0 -> rbfS[0]
      const int  e = tile0 * TM + t;
      const bool v = (tile0 < ntiles) && (e < E);
      const int  s = v ? src[e] : 0;
      const int  d = v ? dst[e] : 0;
      const float dx = R[s * 3 + 0] - R[d * 3 + 0];
      const float dy = R[s * 3 + 1] - R[d * 3 + 1];
      const float dz = R[s * 3 + 2] - R[d * 3 + 2];
      const float dist = sqrtf(dx * dx + dy * dy + dz * dz + 1e-12f);
      const float xx = dist * 0.1f;
      const float x3 = xx * xx * xx, x4 = x3 * xx, x5 = x4 * xx;
      const float cut = (xx < 1.f) ? (1.f - 6.f * x5 + 15.f * x4 - 10.f * x3) : 0.f;
      const float ed  = __expf(-dist);
#pragma unroll
      for (int k = 0; k < 5; ++k) {
        const float ck = 1.0f + (float)k * 0.25f * (EXPM10 - 1.0f);
        const float dv = ed - ck;
        rbfS[t * 5 + k] = cut * __expf(-WIDTH2 * dv * dv);
      }
    }
    const int  e = tile0 * TM + er;
    const bool v = (tile0 < ntiles) && (e < E);
    nextD = -1;
    if (v) {
      const int s = src[e];
      const int d = dst[e];
      nextD = d;
      const float4* ai4 = (const float4*)(aiN + (size_t)d * FDIM + f0);
      const float4* aj4 = (const float4*)(ajN + (size_t)s * FDIM + f0);
#pragma unroll
      for (int q = 0; q < 4; ++q) {
        const float4 a = ai4[q];
        const float4 j = aj4[q];
        aiv[q * 4 + 0] = a.x; aiv[q * 4 + 1] = a.y;
        aiv[q * 4 + 2] = a.z; aiv[q * 4 + 3] = a.w;
        ajv[q * 4 + 0] = j.x; ajv[q * 4 + 1] = j.y;
        ajv[q * 4 + 2] = j.z; ajv[q * 4 + 3] = j.w;
      }
    } else {
#pragma unroll
      for (int j = 0; j < 16; ++j) { aiv[j] = 0.f; ajv[j] = 0.f; }
    }
  }
  __syncthreads();  // k2fS + rbfS[0] visible

  int p = 0;
  for (int tile = blockIdx.x; tile < ntiles; tile += gridDim.x) {
    const int curD = nextD;

    // ---- form m from pipelined registers + rbfS[p] ----
    {
      float rb[5];
#pragma unroll
      for (int k = 0; k < 5; ++k) rb[k] = rbfS[p * TM * 5 + er * 5 + k];
      if (curD >= 0) {
#pragma unroll
        for (int j = 0; j < 16; ++j) {
          const int f = f0 + j;
          float g = 0.f;
#pragma unroll
          for (int k = 0; k < 5; ++k) g += rb[k] * k2fS[f * 5 + k];
          aiS[er * FPS + f] = aiv[j];
          mS [er * FPS + f] = aiv[j] + g * ajv[j];
        }
      } else {
#pragma unroll
        for (int j = 0; j < 16; ++j) {
          aiS[er * FPS + f0 + j] = 0.f;
          mS [er * FPS + f0 + j] = 0.f;
        }
      }
    }

    // ---- stage tile t+gridDim: issue gathers now, consume next iteration ----
    {
      const int ntile = tile + gridDim.x;
      if (t == 0 && ntile + gridDim.x < ntiles) {  // index lines two tiles out
        __builtin_prefetch(src + (ntile + gridDim.x) * TM, 0, 0);
        __builtin_prefetch(dst + (ntile + gridDim.x) * TM, 0, 0);
      }
      if (t < TM) {  // RBF for next tile -> rbfS[p^1]
        const int  e = ntile * TM + t;
        const bool v = (ntile < ntiles) && (e < E);
        const int  s = v ? src[e] : 0;
        const int  d = v ? dst[e] : 0;
        const float dx = R[s * 3 + 0] - R[d * 3 + 0];
        const float dy = R[s * 3 + 1] - R[d * 3 + 1];
        const float dz = R[s * 3 + 2] - R[d * 3 + 2];
        const float dist = sqrtf(dx * dx + dy * dy + dz * dz + 1e-12f);
        const float xx = dist * 0.1f;
        const float x3 = xx * xx * xx, x4 = x3 * xx, x5 = x4 * xx;
        const float cut = (xx < 1.f) ? (1.f - 6.f * x5 + 15.f * x4 - 10.f * x3) : 0.f;
        const float ed  = __expf(-dist);
#pragma unroll
        for (int k = 0; k < 5; ++k) {
          const float ck = 1.0f + (float)k * 0.25f * (EXPM10 - 1.0f);
          const float dv = ed - ck;
          rbfS[(p ^ 1) * TM * 5 + t * 5 + k] = cut * __expf(-WIDTH2 * dv * dv);
        }
      }
      const int  e = ntile * TM + er;
      const bool v = (ntile < ntiles) && (e < E);
      nextD = -1;
      if (v) {
        const int s = src[e];
        const int d = dst[e];
        nextD = d;
        const float4* ai4 = (const float4*)(aiN + (size_t)d * FDIM + f0);
        const float4* aj4 = (const float4*)(ajN + (size_t)s * FDIM + f0);
#pragma unroll
        for (int q = 0; q < 4; ++q) {
          const float4 a = ai4[q];
          const float4 j = aj4[q];
          aiv[q * 4 + 0] = a.x; aiv[q * 4 + 1] = a.y;
          aiv[q * 4 + 2] = a.z; aiv[q * 4 + 3] = a.w;
          ajv[q * 4 + 0] = j.x; ajv[q * 4 + 1] = j.y;
          ajv[q * 4 + 2] = j.z; ajv[q * 4 + 3] = j.w;
        }
      } else {
#pragma unroll
        for (int j = 0; j < 16; ++j) { aiv[j] = 0.f; ajv[j] = 0.f; }
      }
    }
    __syncthreads();   // mS ready (and rbfS[p^1] staged)

    // ---- residual stack on m (2 stages, shared res weights) ----
#pragma unroll
    for (int rs = 0; rs < 2; ++rs)
      residual_stage32(mS, aF, hF, Bf[rs * 2 + 0], Bf[rs * 2 + 1],
                       bs1[rs], bs2[rs], t, lane, ncol, nl, mrow0);

    // ---- new = u * a_i + dense(ssp(m)) ----
    {
#pragma unroll
      for (int j = 0; j < 16; ++j)
        aF[er * APH + f0 + j] = (_Float16)sspf(mS[er * FPS + f0 + j]);
    }
    __syncthreads();
#pragma unroll
    for (int sub = 0; sub < 2; ++sub) {
      v8f c = gemm_tile(aF + sub * 16 * APH, Bf[4], bden, lane);
#pragma unroll
      for (int r = 0; r < 8; ++r) {
        const int idx = (sub * 16 + mrow0 + r) * FPS + ncol + nl;
        mS[idx] = ul * aiS[idx] + c[r];
      }
    }
    __syncthreads();

    // ---- residual stack on new (same weights) ----
#pragma unroll
    for (int rs = 0; rs < 2; ++rs)
      residual_stage32(mS, aF, hF, Bf[rs * 2 + 0], Bf[rs * 2 + 1],
                       bs1[rs], bs2[rs], t, lane, ncol, nl, mrow0);

    // ---- scatter-add messages onto target nodes ----
    if (curD >= 0) {
      float* dp = accum + (size_t)curD * FDIM + f0;
#pragma unroll
      for (int j = 0; j < 16; ++j) atomicAdd(dp + j, mS[er * FPS + f0 + j]);
    }
    __syncthreads();
    p ^= 1;
  }
}

// ---------------------------------------------------------------------------
// node-level: y = ssp(x);  aiN = y@Wdi^T+bi;  ajN = y@Wdj^T+bj
__global__ __launch_bounds__(256, 1)
void node_pre_kernel(const float* __restrict__ x,
                     const float* __restrict__ diW, const float* __restrict__ dib,
                     const float* __restrict__ djW, const float* __restrict__ djb,
                     float* __restrict__ aiN, float* __restrict__ ajN, int N) {
  __shared__ _Float16 aF[16 * APH];
  const int t     = threadIdx.x;
  const int lane  = t & 31;
  const int wave  = t >> 5;
  const int ncol  = wave << 4;
  const int nl    = lane & 15;
  const int mrow0 = (lane >> 4) << 3;

  v16h Bi[4], Bj[4];
#pragma unroll
  for (int kc = 0; kc < 4; ++kc) {
    Bi[kc] = load_bfrag_w(diW, ncol, kc * 32, lane);
    Bj[kc] = load_bfrag_w(djW, ncol, kc * 32, lane);
  }
  const float bi = dib[ncol + nl];
  const float bj = djb[ncol + nl];

  const int ntiles = (N + 15) >> 4;
  for (int tile = blockIdx.x; tile < ntiles; tile += gridDim.x) {
    const int n0 = tile << 4;
    {
      const int er = t >> 4;
      const int f0 = (t & 15) << 3;
      const int n  = n0 + er;
#pragma unroll
      for (int j = 0; j < 8; ++j)
        aF[er * APH + f0 + j] =
            (_Float16)((n < N) ? sspf(x[(size_t)n * FDIM + f0 + j]) : 0.f);
    }
    __syncthreads();
    v8f ci = gemm_tile(aF, Bi, bi, lane);
    v8f cj = gemm_tile(aF, Bj, bj, lane);
#pragma unroll
    for (int r = 0; r < 8; ++r) {
      const int m = mrow0 + r;
      const int n = n0 + m;
      if (n < N) {
        aiN[(size_t)n * FDIM + ncol + nl] = ci[r];
        ajN[(size_t)n * FDIM + ncol + nl] = cj[r];
      }
    }
    __syncthreads();
  }
}

// ---------------------------------------------------------------------------
__global__ void embed_init_kernel(const int* __restrict__ Z,
                                  const float* __restrict__ emb,
                                  float* __restrict__ x, float* __restrict__ mx,
                                  int total) {
  const int i = blockIdx.x * blockDim.x + threadIdx.x;
  if (i < total) {
    const float v = emb[(size_t)Z[i >> 7] * FDIM + (i & 127)];
    x[i]  = v;
    mx[i] = v;
  }
}

__global__ void zero_kernel(float* __restrict__ p, int total) {
  const int i = blockIdx.x * blockDim.x + threadIdx.x;
  if (i < total) p[i] = 0.f;
}

__global__ void update_max_kernel(const float* __restrict__ accum,
                                  float* __restrict__ x, float* __restrict__ mx,
                                  int total) {
  const int i = blockIdx.x * blockDim.x + threadIdx.x;
  if (i < total) {
    const float v = accum[i];
    x[i]  = v;
    mx[i] = fmaxf(mx[i], v);
  }
}

// ---------------------------------------------------------------------------
extern "C" void kernel_launch(void* const* d_in, const int* in_sizes, int n_in,
                              void* d_out, int out_size, void* d_ws, size_t ws_size,
                              hipStream_t stream) {
  (void)n_in; (void)out_size; (void)ws_size;

  const int N = in_sizes[0];
  const int E = in_sizes[2] / 2;

  const int*   Z     = (const int*)d_in[0];
  const float* R     = (const float*)d_in[1];
  const int*   bonds = (const int*)d_in[2];
  const int*   src   = bonds;       // bonds[0] = j (source)
  const int*   dst   = bonds + E;   // bonds[1] = i (target)

  // pytree (sorted-key) flatten: blocks[0..4], then emb.
  // per block: dense.W,b | di.W,b | dj.W,b | k2f.W | res0.d1.W,b | res0.d2.W,b
  //            | res1.d1.W,b | res1.d2.W,b | u     (16 arrays)
  const float* emb = (const float*)d_in[3 + 16 * 5];

  const size_t NF = (size_t)N * FDIM;
  float* x     = (float*)d_ws;
  float* aiN   = x + NF;
  float* ajN   = aiN + NF;
  float* accum = ajN + NF;
  float* mx    = (float*)d_out;

  const int total = (int)NF;
  const int tb = 256;
  const int eb = (total + tb - 1) / tb;

  embed_init_kernel<<<eb, tb, 0, stream>>>(Z, emb, x, mx, total);

  for (int b = 0; b < 5; ++b) {
    const int base = 3 + 16 * b;
    const float* denseW = (const float*)d_in[base + 0];
    const float* denseb = (const float*)d_in[base + 1];
    const float* diW    = (const float*)d_in[base + 2];
    const float* dib    = (const float*)d_in[base + 3];
    const float* djW    = (const float*)d_in[base + 4];
    const float* djb    = (const float*)d_in[base + 5];
    const float* k2fW   = (const float*)d_in[base + 6];
    const float* r0d1W  = (const float*)d_in[base + 7];
    const float* r0d1b  = (const float*)d_in[base + 8];
    const float* r0d2W  = (const float*)d_in[base + 9];
    const float* r0d2b  = (const float*)d_in[base + 10];
    const float* r1d1W  = (const float*)d_in[base + 11];
    const float* r1d1b  = (const float*)d_in[base + 12];
    const float* r1d2W  = (const float*)d_in[base + 13];
    const float* r1d2b  = (const float*)d_in[base + 14];
    const float* u      = (const float*)d_in[base + 15];

    node_pre_kernel<<<1024, 256, 0, stream>>>(x, diW, dib, djW, djb, aiN, ajN, N);
    zero_kernel<<<eb, tb, 0, stream>>>(accum, total);

    EdgeParams P;
    P.k2fW = k2fW;
    P.r0d1W = r0d1W; P.r0d1b = r0d1b; P.r0d2W = r0d2W; P.r0d2b = r0d2b;
    P.r1d1W = r1d1W; P.r1d1b = r1d1b; P.r1d2W = r1d2W; P.r1d2b = r1d2b;
    P.denseW = denseW; P.denseb = denseb; P.u = u;

    edge_kernel<<<2048, 256, 0, stream>>>(aiN, ajN, src, dst, R, P, accum, E);
    update_max_kernel<<<eb, tb, 0, stream>>>(accum, x, mx, total);
  }
}
